// MultiScaleRetention_9285719294238
// MI455X (gfx1250) — compile-verified, matrix operand order checked by reading, not run on hardware
//
#include <hip/hip_runtime.h>
#include <hip/hip_bf16.h>

// ---------------------------------------------------------------------------
// MultiScaleRetention for MI455X (gfx1250, wave32, WMMA f16->f32)
// ---------------------------------------------------------------------------

typedef __attribute__((ext_vector_type(16))) _Float16 v16h;
typedef __attribute__((ext_vector_type(8)))  _Float16 v8h;
typedef __attribute__((ext_vector_type(8)))  float    v8f;

#define D_MODEL 1024
#define NUM_HEADS 4
#define HEAD_QK 256
#define HEAD_V 512
#define SEQ_T 2048
#define BATCH 2
#define NTOK (BATCH * SEQ_T)                   // 4096
#define NCAT (2*D_MODEL + 2*NUM_HEADS*HEAD_V)  // 6144 = q|k|v|g
#define LDY  NCAT

// ---------------------------------------------------------------------------
// Fragment loader: 16x32 f16 tile, row-major memory, CDNA5 WMMA A/B layout
// (lanes 0-15: rows 0-15, K-chunks {0..7,16..23}; lanes 16-31: rows 0-15,
//  K-chunks {8..15,24..31}).  Two 16B loads per lane.
// ---------------------------------------------------------------------------
static __device__ __forceinline__ v16h load_frag16x32(const _Float16* base, int ld) {
    const int lane = threadIdx.x & 31;
    const int row  = lane & 15;
    const int k0   = (lane < 16) ? 0 : 8;
    const _Float16* p = base + (size_t)row * ld + k0;
    v8h a = *(const v8h*)(p);
    v8h b = *(const v8h*)(p + 16);
    v16h f;
#pragma unroll
    for (int i = 0; i < 8; ++i) { f[i] = a[i]; f[i + 8] = b[i]; }
    return f;
}

static __device__ __forceinline__ v8f wmma_f16(v16h a, v16h b, v8f c) {
    return __builtin_amdgcn_wmma_f32_16x16x32_f16(false, a, false, b, (short)0, c,
                                                  false, false);
}

// ---------------------------------------------------------------------------
// Kernel: f32 -> f16 convert (exact sizing, n % 256 == 0)
// ---------------------------------------------------------------------------
__global__ void cvt_f32_to_f16_kernel(const float* __restrict__ src,
                                      _Float16* __restrict__ dst, int n) {
    int i = blockIdx.x * blockDim.x + threadIdx.x;
    if (i < n) dst[i] = (_Float16)src[i];
}

// ---------------------------------------------------------------------------
// Kernel: C[M,N] = A[M,K] @ W[N,K]^T   (A,W f16 row-major; C OutT row-major)
// Block = 128 threads (4 waves); workgroup tile 64x64; wave tile 16x64.
// ---------------------------------------------------------------------------
template <typename OutT>
__global__ void __launch_bounds__(128)
wmma_gemm_nt_kernel(const _Float16* __restrict__ A, int lda,
                    const _Float16* __restrict__ W, int ldw,
                    OutT* __restrict__ C, int ldc, int K) {
    const int wave = threadIdx.x >> 5;
    const int lane = threadIdx.x & 31;
    const int m0 = blockIdx.y * 64 + wave * 16;
    const int n0 = blockIdx.x * 64;

    v8f acc[4] = {};
    for (int k = 0; k < K; k += 32) {
        // prefetch next k-step of the A stream (cold data)
        if (k + 32 < K)
            __builtin_prefetch(A + (size_t)m0 * lda + k + 32, 0, 3);
        v16h af = load_frag16x32(A + (size_t)m0 * lda + k, lda);
#pragma unroll
        for (int j = 0; j < 4; ++j) {
            v16h bf = load_frag16x32(W + (size_t)(n0 + j * 16) * ldw + k, ldw);
            acc[j] = wmma_f16(af, bf, acc[j]);
        }
    }
    // D layout: lanes 0-15 -> col=lane rows 0..7; lanes 16-31 -> col=lane-16 rows 8..15
    const int r0 = (lane < 16) ? 0 : 8;
    const int cn = lane & 15;
#pragma unroll
    for (int j = 0; j < 4; ++j)
#pragma unroll
        for (int r = 0; r < 8; ++r)
            C[(size_t)(m0 + r0 + r) * ldc + (n0 + j * 16 + cn)] = (OutT)acc[j][r];
}

// ---------------------------------------------------------------------------
// Kernel: RoPE in-place on q|k slabs of Y, q additionally scaled by dk^-0.5
// ---------------------------------------------------------------------------
__global__ void rope_kernel(_Float16* __restrict__ Y) {
    const int half = HEAD_QK / 2; // 128
    int idx = blockIdx.x * blockDim.x + threadIdx.x; // NTOK*H*half
    int j   = idx % half;
    int t2  = idx / half;
    int h   = t2 % NUM_HEADS;
    int bt  = t2 / NUM_HEADS;     // 0..4095
    int t   = bt % SEQ_T;

    float inv = __powf(10000.0f, -(float)(2 * j) / (float)HEAD_QK);
    float ang = (float)t * inv;
    float sn, cs;
    __sincosf(ang, &sn, &cs);

    _Float16* qp = Y + (size_t)bt * LDY + h * HEAD_QK;
    _Float16* kp = qp + D_MODEL;

    const float qs = 0.0625f; // 256^-0.5
    float q1 = (float)qp[j], q2 = (float)qp[j + half];
    qp[j]        = (_Float16)((q1 * cs - q2 * sn) * qs);
    qp[j + half] = (_Float16)((q1 * sn + q2 * cs) * qs);

    float k1 = (float)kp[j], k2 = (float)kp[j + half];
    kp[j]        = (_Float16)(k1 * cs - k2 * sn);
    kp[j + half] = (_Float16)(k1 * sn + k2 * cs);
}

// ---------------------------------------------------------------------------
// Kernel: decay-masked retention, flash style.
// grid = (T/16, H, B); block = 128 (4 waves).
// Workgroup: 16 query rows shared by all waves (staged in LDS);
// wave w covers dv columns [w*128, w*128+128).
// Key blocks of 32:  S(16x32) = Q(16x256) @ Kblk^T  (16 WMMAs/wave)
//                    O(16x128) += S @ Vblk          (8 WMMAs/wave)
// Decay is separable: exp(logd*(t-s)) = exp(logd*(t-t0)) * exp(logd*(t0-s)).
// Q lives in LDS (not registers) to keep VGPR pressure spill-free.
// ---------------------------------------------------------------------------
__global__ void __launch_bounds__(128, 1)
retention_kernel(const _Float16* __restrict__ Y, float* __restrict__ O) {
    __shared__ _Float16 sQ[16 * HEAD_QK];   // 8KB: query tile, shared by 4 waves
    __shared__ _Float16 sS[4][16 * 32];     // 4KB: per-wave S staging (D->A relayout)

    const int wave = threadIdx.x >> 5;
    const int lane = threadIdx.x & 31;
    const int t0 = blockIdx.x * 16;
    const int h  = blockIdx.y;
    const int b  = blockIdx.z;

    const int r0 = (lane < 16) ? 0 : 8;     // D-layout row base / frag K base
    const int nn = lane & 15;

    const float logd = logf(1.0f - exp2f(-5.0f - (float)h));
    float rowfac[8];
#pragma unroll
    for (int r = 0; r < 8; ++r) rowfac[r] = __expf(logd * (float)(r0 + r));

    // ---- stage Q (16x256 f16) into LDS: 512 v8h chunks, 4 per thread
    {
        const _Float16* Qg = Y + (size_t)(b * SEQ_T + t0) * LDY + h * HEAD_QK;
#pragma unroll
        for (int i = 0; i < 4; ++i) {
            int c    = threadIdx.x + i * 128;   // chunk id 0..511
            int row  = c >> 5;                  // 32 chunks per row
            int col8 = c & 31;
            *(v8h*)&sQ[c * 8] = *(const v8h*)(Qg + (size_t)row * LDY + col8 * 8);
        }
    }
    __syncthreads();

    const _Float16* Kbase = Y + (size_t)(b * SEQ_T) * LDY + D_MODEL + h * HEAD_QK;
    const _Float16* Vbase = Y + (size_t)(b * SEQ_T) * LDY + 2 * D_MODEL + h * HEAD_V;
    const int c0 = wave * 128;

    v8f oacc[8] = {};

    for (int s0 = 0; s0 <= t0 + 15; s0 += 32) {
        // prefetch next key block rows (cold stream)
        if (s0 + 32 <= t0 + 15)
            __builtin_prefetch(Kbase + (size_t)(s0 + 32 + (lane & 15)) * LDY, 0, 3);

        // ---- S = Q @ Kblk^T : two 16x16 column halves; Q frags from LDS
        v8f sacc[2] = {};
#pragma unroll
        for (int hn = 0; hn < 2; ++hn) {
            const _Float16* kb = Kbase + (size_t)(s0 + hn * 16) * LDY;
#pragma unroll
            for (int kc = 0; kc < 8; ++kc) {
                v16h qf = load_frag16x32(sQ + kc * 32, HEAD_QK);
                v16h kf = load_frag16x32(kb + kc * 32, LDY);
                sacc[hn] = wmma_f16(qf, kf, sacc[hn]);
            }
        }
        // ---- causal decay mask, f32 -> f16, stage into LDS
#pragma unroll
        for (int hn = 0; hn < 2; ++hn) {
            const int s = s0 + hn * 16 + nn;
            const float colfac = __expf(logd * (float)(t0 - s));
#pragma unroll
            for (int r = 0; r < 8; ++r) {
                const int t = t0 + r0 + r;
                float v = (s <= t) ? sacc[hn][r] * rowfac[r] * colfac : 0.0f;
                sS[wave][(r0 + r) * 32 + hn * 16 + nn] = (_Float16)v;
            }
        }
        // ---- reload S as an A-operand fragment (same wave: dscnt ordering)
        const _Float16* sp = &sS[wave][nn * 32 + r0];
        v8h a0 = *(const v8h*)sp;
        v8h a1 = *(const v8h*)(sp + 16);
        v16h sf;
#pragma unroll
        for (int i = 0; i < 8; ++i) { sf[i] = a0[i]; sf[i + 8] = a1[i]; }

        // ---- O += S @ Vblk   (V fragment: per-lane column gather, coalesced)
#pragma unroll
        for (int j = 0; j < 8; ++j) {
            const _Float16* vp = Vbase + c0 + j * 16 + nn;
            v16h vf;
#pragma unroll
            for (int e = 0; e < 8; ++e) {
                vf[e]     = vp[(size_t)(s0 + r0 + e) * LDY];
                vf[e + 8] = vp[(size_t)(s0 + r0 + 16 + e) * LDY];
            }
            oacc[j] = wmma_f16(sf, vf, oacc[j]);
        }
    }

    // ---- store O tile (f32), layout (token, h*dv + c)
    float* ob = O + (size_t)(b * SEQ_T + t0) * (NUM_HEADS * HEAD_V) + h * HEAD_V + c0;
#pragma unroll
    for (int j = 0; j < 8; ++j)
#pragma unroll
        for (int r = 0; r < 8; ++r)
            ob[(size_t)(r0 + r) * (NUM_HEADS * HEAD_V) + j * 16 + nn] = oacc[j][r];
}

// ---------------------------------------------------------------------------
// Kernel: RMS norm over dv + SiLU gate.  grid=(NTOK, H), block=128.
// ---------------------------------------------------------------------------
__global__ void normgate_kernel(const float* __restrict__ O,
                                const _Float16* __restrict__ Y,
                                const float* __restrict__ gw,
                                _Float16* __restrict__ OG) {
    __shared__ float red[128];
    const int bt = blockIdx.x;
    const int h  = blockIdx.y;
    const float* op = O + (size_t)bt * (NUM_HEADS * HEAD_V) + h * HEAD_V;

    float4 v4 = ((const float4*)op)[threadIdx.x];
    float vals[4] = {v4.x, v4.y, v4.z, v4.w};
    float s = vals[0]*vals[0] + vals[1]*vals[1] + vals[2]*vals[2] + vals[3]*vals[3];
    red[threadIdx.x] = s;
    __syncthreads();
    for (int off = 64; off > 0; off >>= 1) {
        if (threadIdx.x < off) red[threadIdx.x] += red[threadIdx.x + off];
        __syncthreads();
    }
    const float rms = rsqrtf(red[0] / (float)HEAD_V + 1e-5f);

    const _Float16* gp = Y + (size_t)bt * LDY + 2 * D_MODEL + NUM_HEADS * HEAD_V + h * HEAD_V;
    _Float16* ogp = OG + (size_t)bt * (NUM_HEADS * HEAD_V) + h * HEAD_V;
    const int c = threadIdx.x * 4;
#pragma unroll
    for (int i = 0; i < 4; ++i) {
        float g  = (float)gp[c + i];
        float on = vals[i] * rms * gw[c + i];
        float sg = g / (1.0f + __expf(-g));   // g * sigmoid(g)
        ogp[c + i] = (_Float16)(on * sg);
    }
}

// ---------------------------------------------------------------------------
// Launch
// ---------------------------------------------------------------------------
extern "C" void kernel_launch(void* const* d_in, const int* in_sizes, int n_in,
                              void* d_out, int out_size, void* d_ws, size_t ws_size,
                              hipStream_t stream) {
    const float* x   = (const float*)d_in[0];
    const float* Wq  = (const float*)d_in[1];
    const float* Wk  = (const float*)d_in[2];
    const float* Wv  = (const float*)d_in[3];
    const float* Wg  = (const float*)d_in[4];
    const float* Wo  = (const float*)d_in[5];
    const float* gnw = (const float*)d_in[6];
    float* out = (float*)d_out;

    // workspace carve (all offsets 256B-aligned)
    char* ws = (char*)d_ws;
    _Float16* xh   = (_Float16*)ws; ws += (size_t)NTOK * D_MODEL * 2;               // 8 MB
    _Float16* Wcat = (_Float16*)ws; ws += (size_t)NCAT * D_MODEL * 2;               // 12 MB
    _Float16* Woh  = (_Float16*)ws; ws += (size_t)D_MODEL * (NUM_HEADS*HEAD_V) * 2; // 4 MB
    _Float16* Ycat = (_Float16*)ws; ws += (size_t)NTOK * NCAT * 2;                  // 48 MB
    float*    Obuf = (float*)ws;    ws += (size_t)NTOK * NUM_HEADS * HEAD_V * 4;    // 32 MB
    _Float16* OG   = (_Float16*)ws; ws += (size_t)NTOK * NUM_HEADS * HEAD_V * 2;    // 16 MB

    // 1) f32 -> f16 conversions (weights packed into Wcat = [Wq;Wk;Wv;Wg])
    {
        const int TB = 256;
        int nx = NTOK * D_MODEL;
        cvt_f32_to_f16_kernel<<<nx / TB, TB, 0, stream>>>(x, xh, nx);
        int nqk = D_MODEL * D_MODEL;               // 1,048,576
        int nvg = NUM_HEADS * HEAD_V * D_MODEL;    // 2,097,152
        cvt_f32_to_f16_kernel<<<nqk / TB, TB, 0, stream>>>(Wq, Wcat, nqk);
        cvt_f32_to_f16_kernel<<<nqk / TB, TB, 0, stream>>>(Wk, Wcat + (size_t)nqk, nqk);
        cvt_f32_to_f16_kernel<<<nvg / TB, TB, 0, stream>>>(Wv, Wcat + (size_t)2 * nqk, nvg);
        cvt_f32_to_f16_kernel<<<nvg / TB, TB, 0, stream>>>(Wg, Wcat + (size_t)2 * nqk + nvg, nvg);
        cvt_f32_to_f16_kernel<<<nvg / TB, TB, 0, stream>>>(Wo, Woh, nvg);
    }

    // 2) fused projection GEMM: Ycat[4096, 6144] = xh @ Wcat^T
    {
        dim3 grid(NCAT / 64, NTOK / 64);
        wmma_gemm_nt_kernel<_Float16><<<grid, 128, 0, stream>>>(
            xh, D_MODEL, Wcat, D_MODEL, Ycat, LDY, D_MODEL);
    }

    // 3) RoPE (in-place on q|k slabs, q scaled by dk^-0.5)
    {
        int n = NTOK * NUM_HEADS * (HEAD_QK / 2); // 2,097,152
        rope_kernel<<<n / 256, 256, 0, stream>>>(Ycat);
    }

    // 4) decay-masked retention
    {
        dim3 grid(SEQ_T / 16, NUM_HEADS, BATCH);
        retention_kernel<<<grid, 128, 0, stream>>>(Ycat, Obuf);
    }

    // 5) RMS norm + SiLU gate
    {
        dim3 grid(NTOK, NUM_HEADS);
        normgate_kernel<<<grid, 128, 0, stream>>>(Obuf, Ycat, gnw, OG);
    }

    // 6) output projection: out[4096, 1024] = OG @ Wo^T  (f32 store)
    {
        dim3 grid(D_MODEL / 64, NTOK / 64);
        wmma_gemm_nt_kernel<float><<<grid, 128, 0, stream>>>(
            OG, NUM_HEADS * HEAD_V, Woh, NUM_HEADS * HEAD_V, out, D_MODEL,
            NUM_HEADS * HEAD_V);
    }
}
